// DAGNN_14491219657204
// MI455X (gfx1250) — matrix operand order repaired and checked by reading, not compiled, not run on hardware
//
#include <hip/hip_runtime.h>
#include <hip/hip_bf16.h>
#include <cstdint>

typedef __attribute__((ext_vector_type(16))) _Float16 v16h;
typedef __attribute__((ext_vector_type(8)))  _Float16 v8h;
typedef __attribute__((ext_vector_type(8)))  float    v8f;
typedef __attribute__((ext_vector_type(4)))  unsigned int v4u;
typedef __attribute__((ext_vector_type(8)))  unsigned int v8u;

#define N_NODES 100000
#define N_EDGES 1600000
#define NN_IN   512
#define NN_HID  256
#define NN_OUT  64
#define HOPS    10
#define HSTRIDE ((HOPS + 1) * NN_OUT)           // 704 floats per node in H
#define H_BYTES ((size_t)N_NODES * HSTRIDE * 4) // 281.6 MB

// ---------------------------------------------------------------------------
// Kernel 0: zero the H workspace so hop scatter-adds start from zero.
// ---------------------------------------------------------------------------
__global__ __launch_bounds__(256) void dagnn_zero(float4* __restrict__ p, long long n4) {
    long long i = (long long)blockIdx.x * blockDim.x + threadIdx.x;
    if (i < n4) p[i] = make_float4(0.f, 0.f, 0.f, 0.f);
}

// ---------------------------------------------------------------------------
// Kernel 0b: one-time f32 -> f16 weight conversion (W1h, W2h in workspace).
// ---------------------------------------------------------------------------
__global__ __launch_bounds__(256) void dagnn_cvt_f16(const float* __restrict__ src,
                                                     _Float16* __restrict__ dst, int n) {
    int i = blockIdx.x * blockDim.x + threadIdx.x;
    if (i < n) dst[i] = (_Float16)src[i];
}

// ---------------------------------------------------------------------------
// TDM: one tensor_load_to_lds moving a [256 rows x 32 halves] f16 tile of W1h
// (row stride 512 halves) into LDS at ldsoff. D# per CDNA5 ISA:
//   group0: count=1 | lds_addr | global_addr(57b) | type=2
//   group1: data_size=2B, tensor 512x256, tile 32x256, dim0_stride=512
//   groups 2/3: zero (2-D tensor)
// TENSOR ops ignore EXEC and issue once per executing wave -> gate on wave 0.
// ---------------------------------------------------------------------------
__device__ __forceinline__ void tdm_load_w1_slice(const _Float16* gsrc, unsigned ldsoff) {
    unsigned long long ga = (unsigned long long)(uintptr_t)gsrc;
    v4u g0 = { 1u,                                   // count=1, user descriptor
               ldsoff,                               // LDS byte address
               (unsigned)ga,                         // global_addr[31:0]
               ((unsigned)(ga >> 32) & 0x01FFFFFFu) | (2u << 30) }; // [56:32], type=2
    v8u g1 = { 0x00010000u,                          // wg_mask=0, data_size=1 (2B)
               (unsigned)(NN_IN  << 16),             // tensor_dim0 = 512
               (unsigned)(NN_HID << 16),             // tensor_dim1 = 256
               (32u << 16),                          // tile_dim0 = 32
               (unsigned)NN_HID,                     // tile_dim1 = 256, tile_dim2 = 0
               (unsigned)NN_IN,                      // tensor_dim0_stride = 512
               0u, 0u };
    v4u gz = { 0u, 0u, 0u, 0u };
    asm volatile("tensor_load_to_lds %0, %1, %2, %3"
                 :: "s"(g0), "s"(g1), "s"(gz), "s"(gz)
                 : "memory");
}

// ---------------------------------------------------------------------------
// Kernel 1: fused MLP  h0 = relu(x@W1^T + b1) @ W2^T + b2  ->  H[:, 0, :]
// 128 threads = 4 waves; each wave owns a 16-row tile of x.
// GEMM1: W1h k-slices DMA'd into double-buffered LDS by the TDM (pipelined one
// slice ahead, s_wait_tensorcnt + barrier publish). B fragments stream through
// two disjoint 4-fragment register groups so each burst of 4 WMMAs overlaps
// the next group's 8 ds_load_b128.
// GEMM2: h1 tile staged in LDS (unioned with the W buffers), W2h fragments
// loaded straight from global as f16.
// ---------------------------------------------------------------------------
__global__ __launch_bounds__(128) void dagnn_mlp_wmma(
    const float* __restrict__ x,   const _Float16* __restrict__ W1h,
    const float* __restrict__ b1,  const _Float16* __restrict__ W2h,
    const float* __restrict__ b2,  float* __restrict__ H)
{
    __shared__ union {
        _Float16 w[2][NN_HID * 32];      // 2 x 16 KB W1 slices (GEMM1 phase)
        _Float16 h1[4][16 * NN_HID];     // 4 x 8 KB per-wave h1 tiles (GEMM2 phase)
    } lds;                               // 32 KB total

    const int tid   = threadIdx.x;
    const int lane  = tid & 31;
    const int wave  = tid >> 5;
    const int l16   = lane & 15;
    const int lhalf = lane >> 4;                  // 0 or 1
    const int tile  = blockIdx.x * 4 + wave;      // 16-row tile index
    const bool active = (tile < N_NODES / 16);
    const int row0  = tile * 16;

    v8f zero8 = {0.f, 0.f, 0.f, 0.f, 0.f, 0.f, 0.f, 0.f};
    v8f acc1[16];
    #pragma unroll
    for (int i = 0; i < 16; ++i) acc1[i] = zero8;

    const unsigned ldsWBase = (unsigned)(uintptr_t)&lds.w[0][0];

    // Prologue: DMA slice 0 into buffer 0.
    if (wave == 0) tdm_load_w1_slice(W1h, ldsWBase);

    // ---- GEMM1: K loop over 512 in steps of 32, TDM pipelined 1 deep ----
    for (int k0 = 0; k0 < NN_IN; k0 += 32) {
        const int buf = (k0 >> 5) & 1;
        if (wave == 0) __builtin_amdgcn_s_wait_tensorcnt(0);  // slice `buf` landed
        __syncthreads();                                      // publish to all waves
        if (wave == 0 && (k0 + 32) < NN_IN)                   // prefetch next slice
            tdm_load_w1_slice(W1h + (k0 + 32), ldsWBase + (buf ^ 1) * (NN_HID * 32 * 2));

        if (active) {
            // A fragment: M = row0 + l16; K = k0 + lhalf*16 + [0..15].
            const float* arow = x + (size_t)(row0 + l16) * NN_IN + k0 + lhalf * 16;
            if (k0 + 32 < NN_IN) __builtin_prefetch(arow + 32, 0, 0);
            float4 c0 = *(const float4*)(arow);
            float4 c1 = *(const float4*)(arow + 4);
            float4 c2 = *(const float4*)(arow + 8);
            float4 c3 = *(const float4*)(arow + 12);
            v16h a;
            a[0]=(_Float16)c0.x; a[1]=(_Float16)c0.y; a[2]=(_Float16)c0.z; a[3]=(_Float16)c0.w;
            a[4]=(_Float16)c1.x; a[5]=(_Float16)c1.y; a[6]=(_Float16)c1.z; a[7]=(_Float16)c1.w;
            a[8]=(_Float16)c2.x; a[9]=(_Float16)c2.y; a[10]=(_Float16)c2.z; a[11]=(_Float16)c2.w;
            a[12]=(_Float16)c3.x; a[13]=(_Float16)c3.y; a[14]=(_Float16)c3.z; a[15]=(_Float16)c3.w;

            // B fragments: N = nt*16 + l16, K = lhalf*16 + h. Fragment nt lives
            // at wb + nt*512 halves. Two disjoint 4-fragment groups, ping-pong:
            // group g+1's loads are issued before group g's 4 WMMAs.
            const _Float16* wb = &lds.w[buf][l16 * 32 + lhalf * 16];
            v16h bA[4], bB[4];
            #pragma unroll
            for (int j = 0; j < 4; ++j)
                bA[j] = *(const v16h*)(wb + j * 512);
            #pragma unroll
            for (int g = 0; g < 4; ++g) {
                v16h* cur = ((g & 1) == 0) ? bA : bB;
                v16h* nxt = ((g & 1) == 0) ? bB : bA;
                if (g < 3) {
                    #pragma unroll
                    for (int j = 0; j < 4; ++j)
                        nxt[j] = *(const v16h*)(wb + ((g + 1) * 4 + j) * 512);
                }
                #pragma unroll
                for (int j = 0; j < 4; ++j)
                    acc1[g * 4 + j] = __builtin_amdgcn_wmma_f32_16x16x32_f16(
                        false, a, false, cur[j], (short)0, acc1[g * 4 + j], false, false);
            }
        }
    }

    // All waves must be done reading lds.w before h1 overlays it.
    __syncthreads();
    if (!active) return;

    // ---- bias + ReLU, stage h1 tile (16x256 f16) into this wave's LDS slot ----
    #pragma unroll
    for (int nt = 0; nt < 16; ++nt) {
        float bias = b1[nt * 16 + l16];
        #pragma unroll
        for (int r = 0; r < 8; ++r) {
            float v = acc1[nt][r] + bias;
            v = v > 0.f ? v : 0.f;
            // C layout: M = r + lhalf*8, N = nt*16 + l16
            lds.h1[wave][(r + lhalf * 8) * NN_HID + nt * 16 + l16] = (_Float16)v;
        }
    }
    // same-wave LDS RAW below: DS ops are in-order per wave; compiler waits.

    // ---- GEMM2: 16x64, K=256 in steps of 32; W2h fragments direct f16 ----
    v8f acc2[4];
    #pragma unroll
    for (int i = 0; i < 4; ++i) acc2[i] = zero8;

    #pragma unroll
    for (int k0 = 0; k0 < NN_HID; k0 += 32) {
        v8h lo = *(const v8h*)&lds.h1[wave][l16 * NN_HID + k0 + lhalf * 16];
        v8h hi = *(const v8h*)&lds.h1[wave][l16 * NN_HID + k0 + lhalf * 16 + 8];
        v16h a;
        #pragma unroll
        for (int j = 0; j < 8; ++j) { a[j] = lo[j]; a[j + 8] = hi[j]; }

        #pragma unroll
        for (int nt = 0; nt < 4; ++nt) {
            v16h b = *(const v16h*)(W2h + (size_t)(nt * 16 + l16) * NN_HID
                                        + k0 + lhalf * 16);
            acc2[nt] = __builtin_amdgcn_wmma_f32_16x16x32_f16(
                false, a, false, b, (short)0, acc2[nt], false, false);
        }
    }

    // ---- epilogue: + b2, store hop-0 slice into H[node, 0, :] ----
    #pragma unroll
    for (int nt = 0; nt < 4; ++nt) {
        float bias = b2[nt * 16 + l16];
        #pragma unroll
        for (int r = 0; r < 8; ++r) {
            int node = row0 + r + lhalf * 8;
            H[(size_t)node * HSTRIDE + nt * 16 + l16] = acc2[nt][r] + bias;
        }
    }
}

// ---------------------------------------------------------------------------
// Kernel 2: one SpMM hop. One wave per edge; lane holds float2 of the 64-wide
// feature row (256 B coalesced gather, global_atomic_add_f32 scatter).
// ---------------------------------------------------------------------------
__global__ __launch_bounds__(256) void dagnn_spmm_hop(
    const int* __restrict__ erow, const int* __restrict__ ecol,
    const float* __restrict__ evals, float* __restrict__ H, int hop)
{
    int gw   = (int)((blockIdx.x * blockDim.x + threadIdx.x) >> 5);
    int lane = threadIdx.x & 31;
    if (gw >= N_EDGES) return;
    int   r = erow[gw];
    int   c = ecol[gw];
    float v = evals[gw];
    const float* src = H + (size_t)c * HSTRIDE + (hop - 1) * NN_OUT + lane * 2;
    float2 h = *(const float2*)src;
    float* dst = H + (size_t)r * HSTRIDE + hop * NN_OUT + lane * 2;
    unsafeAtomicAdd(dst,     v * h.x);
    unsafeAtomicAdd(dst + 1, v * h.y);
}

// ---------------------------------------------------------------------------
// Kernel 3: hop attention. One wave per node; lane owns 2 features.
// ---------------------------------------------------------------------------
__global__ __launch_bounds__(256) void dagnn_attn(
    const float* __restrict__ H, const float* __restrict__ s,
    float* __restrict__ out)
{
    int gw   = (int)((blockIdx.x * blockDim.x + threadIdx.x) >> 5);
    int lane = threadIdx.x & 31;
    if (gw >= N_NODES) return;
    float sx = s[lane * 2];
    float sy = s[lane * 2 + 1];
    const float* hrow = H + (size_t)gw * HSTRIDE + lane * 2;
    float ax = 0.f, ay = 0.f;
    #pragma unroll
    for (int k = 0; k <= HOPS; ++k) {
        float2 h = *(const float2*)(hrow + k * NN_OUT);
        float p = h.x * sx + h.y * sy;
        #pragma unroll
        for (int off = 16; off > 0; off >>= 1)
            p += __shfl_xor(p, off, 32);
        float sg = 1.f / (1.f + __expf(-p));
        ax += sg * h.x;
        ay += sg * h.y;
    }
    float2* o = (float2*)(out + (size_t)gw * NN_OUT + lane * 2);
    *o = make_float2(ax, ay);
}

// ---------------------------------------------------------------------------
extern "C" void kernel_launch(void* const* d_in, const int* in_sizes, int n_in,
                              void* d_out, int out_size, void* d_ws, size_t ws_size,
                              hipStream_t stream) {
    const float* x    = (const float*)d_in[0];
    const int*   erow = (const int*)d_in[1];
    const int*   ecol = (const int*)d_in[2];
    const float* ev   = (const float*)d_in[3];
    const float* W1   = (const float*)d_in[4];
    const float* b1   = (const float*)d_in[5];
    const float* W2   = (const float*)d_in[6];
    const float* b2   = (const float*)d_in[7];
    const float* s    = (const float*)d_in[8];

    float*    H   = (float*)d_ws;                            // [N, 11, 64] f32
    _Float16* W1h = (_Float16*)((char*)d_ws + H_BYTES);      // 256 KB f16
    _Float16* W2h = W1h + NN_HID * NN_IN;                    // 32 KB f16
    float*    out = (float*)d_out;                           // [N, 64]

    // 0) zero H; convert weights to f16 once
    long long n4 = (long long)N_NODES * HSTRIDE / 4;
    dagnn_zero<<<(unsigned)((n4 + 255) / 256), 256, 0, stream>>>((float4*)H, n4);
    dagnn_cvt_f16<<<(NN_HID * NN_IN + 255) / 256, 256, 0, stream>>>(W1, W1h, NN_HID * NN_IN);
    dagnn_cvt_f16<<<(NN_OUT * NN_HID + 255) / 256, 256, 0, stream>>>(W2, W2h, NN_OUT * NN_HID);

    // 1) fused WMMA MLP (TDM-staged weights) -> H[:, 0, :]
    int nTiles = N_NODES / 16;                               // 6250
    dagnn_mlp_wmma<<<(nTiles + 3) / 4, 128, 0, stream>>>(x, W1h, b1, W2h, b2, H);

    // 2) 10 propagation hops
    int spmmBlocks = (int)(((long long)N_EDGES * 32 + 255) / 256);
    for (int hop = 1; hop <= HOPS; ++hop)
        dagnn_spmm_hop<<<spmmBlocks, 256, 0, stream>>>(erow, ecol, ev, H, hop);

    // 3) sigmoid hop-attention -> out
    int attnBlocks = (int)(((long long)N_NODES * 32 + 255) / 256);
    dagnn_attn<<<attnBlocks, 256, 0, stream>>>(H, s, out);
}
